// SDFDiscriminator_91328184582508
// MI455X (gfx1250) — matrix operand order repaired and checked by reading, not compiled, not run on hardware
//
#include <hip/hip_runtime.h>
#include <hip/hip_bf16.h>

typedef _Float16 v16h __attribute__((ext_vector_type(16)));
typedef _Float16 v8h  __attribute__((ext_vector_type(8)));
typedef float    v8f  __attribute__((ext_vector_type(8)));

// ---------------------------------------------------------------------------
// Farthest point sampling: single persistent workgroup, distances in LDS.
// Matches jnp.argmax first-max tie-breaking (value desc, index asc).
// ---------------------------------------------------------------------------
__global__ __launch_bounds__(1024) void fps_kernel(const float* __restrict__ pos, int n, int m,
                                                   int* __restrict__ idx, float* __restrict__ posy) {
  __shared__ float d[8192];
  __shared__ float rv[1024];
  __shared__ int   ri[1024];
  __shared__ float cx, cy, cz;
  const int t = threadIdx.x, T = blockDim.x;
  const float p0x = pos[0], p0y = pos[1], p0z = pos[2];
  for (int j = t; j < n; j += T) {
    float dx = pos[j*3] - p0x, dy = pos[j*3+1] - p0y, dz = pos[j*3+2] - p0z;
    d[j] = dx*dx + dy*dy + dz*dz;
  }
  if (t == 0) { idx[0] = 0; posy[0] = p0x; posy[1] = p0y; posy[2] = p0z; }
  __syncthreads();
  for (int s = 1; s < m; ++s) {
    float bv = -1.0f; int bi = n;
    for (int j = t; j < n; j += T) { float v = d[j]; if (v > bv) { bv = v; bi = j; } }
    rv[t] = bv; ri[t] = bi;
    __syncthreads();
    for (int off = T >> 1; off > 0; off >>= 1) {
      if (t < off) {
        float v2 = rv[t+off]; int i2 = ri[t+off];
        if (v2 > rv[t] || (v2 == rv[t] && i2 < ri[t])) { rv[t] = v2; ri[t] = i2; }
      }
      __syncthreads();
    }
    if (t == 0) {
      int i = ri[0]; idx[s] = i;
      cx = pos[i*3]; cy = pos[i*3+1]; cz = pos[i*3+2];
      posy[s*3] = cx; posy[s*3+1] = cy; posy[s*3+2] = cz;
    }
    __syncthreads();
    float qx = cx, qy = cy, qz = cz;
    for (int j = t; j < n; j += T) {
      float dx = pos[j*3] - qx, dy = pos[j*3+1] - qy, dz = pos[j*3+2] - qz;
      float v = dx*dx + dy*dy + dz*dz;
      if (v < d[j]) d[j] = v;
    }
    __syncthreads();
  }
}

// ---------------------------------------------------------------------------
// Radius neighbors: one wave32 per query; ballot + prefix-popcount compaction
// collects the first 64 in-radius indices in ascending order (== reference
// top_k(-keyi) semantics). Unfilled slots set to 0, valid count recorded.
// ---------------------------------------------------------------------------
__global__ __launch_bounds__(32) void radius_kernel(const float* __restrict__ posx, int n,
                                                    const float* __restrict__ posy,
                                                    int* __restrict__ nb, int* __restrict__ cnt,
                                                    float r2) {
  const int q = blockIdx.x, lane = threadIdx.x;
  const float qx = posy[q*3], qy = posy[q*3+1], qz = posy[q*3+2];
  int* out = nb + (size_t)q * 64;
  int count = 0;
  for (int base = 0; base < n && count < 64; base += 32) {
    int j = base + lane;
    bool pred = false;
    if (j < n) {
      float dx = posx[j*3] - qx, dy = posx[j*3+1] - qy, dz = posx[j*3+2] - qz;
      pred = (dx*dx + dy*dy + dz*dz) <= r2;
    }
    unsigned msk = __builtin_amdgcn_ballot_w32(pred);
    if (pred) {
      int slot = count + __popc(msk & ((1u << lane) - 1u));
      if (slot < 64) out[slot] = j;
    }
    count += __popc(msk);
  }
  if (count > 64) count = 64;
  for (int s = count + lane; s < 64; s += 32) out[s] = 0;
  if (lane == 0) cnt[q] = count;
}

// ---------------------------------------------------------------------------
// Gather/concat kernels build zero-padded f16 activation matrices (K mult 32)
// ---------------------------------------------------------------------------
__global__ void gather1_kernel(const float* __restrict__ feat, const float* __restrict__ pos,
                               const int* __restrict__ nb, const float* __restrict__ posy,
                               _Float16* __restrict__ A, int M) {
  int row = blockIdx.x * blockDim.x + threadIdx.x;
  if (row >= M * 64) return;
  int q = row >> 6, j = nb[row];
  _Float16* a = A + (size_t)row * 32;
  a[0] = (_Float16)feat[j];
  a[1] = (_Float16)(pos[j*3]   - posy[q*3]);
  a[2] = (_Float16)(pos[j*3+1] - posy[q*3+1]);
  a[3] = (_Float16)(pos[j*3+2] - posy[q*3+2]);
#pragma unroll
  for (int c = 4; c < 32; ++c) a[c] = (_Float16)0.0f;
}

__global__ void gather2_kernel(const _Float16* __restrict__ x1, const float* __restrict__ posx,
                               const int* __restrict__ nb, const float* __restrict__ posy,
                               _Float16* __restrict__ A, int M) {
  int row = blockIdx.x * blockDim.x + threadIdx.x;
  if (row >= M * 64) return;
  int q = row >> 6, j = nb[row];
  _Float16* a = A + (size_t)row * 160;
  const _Float16* src = x1 + (size_t)j * 128;
  for (int c = 0; c < 128; ++c) a[c] = src[c];
  a[128] = (_Float16)(posx[j*3]   - posy[q*3]);
  a[129] = (_Float16)(posx[j*3+1] - posy[q*3+1]);
  a[130] = (_Float16)(posx[j*3+2] - posy[q*3+2]);
  for (int c = 131; c < 160; ++c) a[c] = (_Float16)0.0f;
}

__global__ void concat3_kernel(const _Float16* __restrict__ x2, const float* __restrict__ pos2,
                               _Float16* __restrict__ A) {
  int row = blockIdx.x * blockDim.x + threadIdx.x;
  if (row >= 1024) return;
  _Float16* a = A + (size_t)row * 288;
  const _Float16* src = x2 + (size_t)row * 256;
  for (int c = 0; c < 256; ++c) a[c] = src[c];
  a[256] = (_Float16)pos2[row*3];
  a[257] = (_Float16)pos2[row*3+1];
  a[258] = (_Float16)pos2[row*3+2];
  for (int c = 259; c < 288; ++c) a[c] = (_Float16)0.0f;
}

// Weight f32 (Cout x Cin) -> f16 (Cout x Kp), K zero-padded
__global__ void convw_kernel(const float* __restrict__ W, _Float16* __restrict__ Wh,
                             int Cout, int Cin, int Kp) {
  int t = blockIdx.x * blockDim.x + threadIdx.x;
  if (t >= Cout * Kp) return;
  int co = t / Kp, k = t % Kp;
  Wh[t] = (_Float16)((k < Cin) ? W[(size_t)co * Cin + k] : 0.0f);
}

// ---------------------------------------------------------------------------
// WMMA GEMM: Y = epilogue(A @ Wh^T). One wave = one 16x16 tile, K step 32.
// A: (R x Kp) f16 row-major; Wh: (Cout x Kp) f16 row-major (rows of W are
// columns of B). Per-lane loads match the CDNA5 wave32 VGPR striping:
//   A lane: m=lane&15, half=lane>>4 -> v0..3 = K[k0+8h .. +7], v4..7 = +16
//   B lane: n=lane&15              -> v0..3 = K[k0+8h .. +7], v4..7 = +16
// Epilogue: +bias, ReLU, optional inference BatchNorm; all channel-indexed
// so constant per lane. Output f16 (feeds next layer's A).
// ---------------------------------------------------------------------------
__global__ __launch_bounds__(32) void gemm_wmma(const _Float16* __restrict__ A,
                                                const _Float16* __restrict__ Wh,
                                                const float* __restrict__ bias,
                                                const float* __restrict__ gamma,
                                                const float* __restrict__ beta,
                                                const float* __restrict__ mu,
                                                const float* __restrict__ var,
                                                _Float16* __restrict__ Y,
                                                int Kp, int Cout, int use_bn) {
  const int lane = threadIdx.x;
  const int row0 = blockIdx.x * 16;
  const int n0   = blockIdx.y * 16;
  const int half = lane >> 4;
  const int mr   = lane & 15;
  const _Float16* arow = A  + (size_t)(row0 + mr) * Kp;
  const _Float16* brow = Wh + (size_t)(n0  + mr) * Kp;   // n == lane&15 == mr
  v8f acc = {};
  for (int k0 = 0; k0 < Kp; k0 += 32) {
    v16h a, b;
    *((v8h*)&a)       = *(const v8h*)(arow + k0 + half * 8);
    *(((v8h*)&a) + 1) = *(const v8h*)(arow + k0 + 16 + half * 8);
    *((v8h*)&b)       = *(const v8h*)(brow + k0 + half * 8);
    *(((v8h*)&b) + 1) = *(const v8h*)(brow + k0 + 16 + half * 8);
    acc = __builtin_amdgcn_wmma_f32_16x16x32_f16(false, a, false, b, (short)0, acc, false, false);
  }
  const int co = n0 + mr;
  const float bb = bias[co];
  float mm = 0.0f, inv = 1.0f, gg = 1.0f, be = 0.0f;
  if (use_bn) { mm = mu[co]; inv = rsqrtf(var[co] + 1e-5f); gg = gamma[co]; be = beta[co]; }
#pragma unroll
  for (int r = 0; r < 8; ++r) {
    float y = acc[r] + bb;
    y = fmaxf(y, 0.0f);
    if (use_bn) y = (y - mm) * inv * gg + be;
    Y[(size_t)(row0 + r + half * 8) * Cout + co] = (_Float16)y;
  }
}

// Masked max over 64 neighbors (only first cnt[q] valid; self always valid)
__global__ void maxpool_kernel(const _Float16* __restrict__ Yin, const int* __restrict__ cnt,
                               _Float16* __restrict__ X, int M, int C) {
  int t = blockIdx.x * blockDim.x + threadIdx.x;
  if (t >= M * C) return;
  int q = t / C, c = t % C;
  int n = cnt[q];
  const _Float16* p = Yin + (size_t)q * 64 * C + c;
  float best = -3.0e38f;
  for (int i = 0; i < n; ++i) best = fmaxf(best, (float)p[(size_t)i * C]);
  X[t] = (_Float16)best;
}

// Global max over 1024 rows -> row 0 of a 16x1024 f16 A-tile (rows 1..15 = 0)
__global__ void gmax_kernel(const _Float16* __restrict__ Yin, _Float16* __restrict__ Afc) {
  int c = blockIdx.x * blockDim.x + threadIdx.x;
  if (c >= 1024) return;
  float best = -3.0e38f;
  for (int r = 0; r < 1024; ++r) best = fmaxf(best, (float)Yin[(size_t)r * 1024 + c]);
  Afc[c] = (_Float16)best;
  for (int r = 1; r < 16; ++r) Afc[(size_t)r * 1024 + c] = (_Float16)0.0f;
}

// fc3: out = sigmoid(dot(h[256], W) + b), single wave
__global__ __launch_bounds__(32) void fc3_kernel(const _Float16* __restrict__ h,
                                                 const float* __restrict__ W,
                                                 const float* __restrict__ b,
                                                 float* __restrict__ out) {
  __shared__ float red[32];
  int l = threadIdx.x;
  float s = 0.0f;
  for (int k = l; k < 256; k += 32) s += (float)h[k] * W[k];
  red[l] = s;
  __syncthreads();
  if (l == 0) {
    float t = 0.0f;
    for (int i = 0; i < 32; ++i) t += red[i];
    t += b[0];
    out[0] = 1.0f / (1.0f + expf(-t));
  }
}

// ---------------------------------------------------------------------------
// Host orchestration
// ---------------------------------------------------------------------------
extern "C" void kernel_launch(void* const* d_in, const int* in_sizes, int n_in,
                              void* d_out, int out_size, void* d_ws, size_t ws_size,
                              hipStream_t stream) {
  (void)in_sizes; (void)n_in; (void)out_size; (void)ws_size;
  char* ws = (char*)d_ws;
  size_t off = 0;
  auto take = [&](size_t bytes) -> char* {
    char* p = ws + off;
    off = (off + bytes + 255) & ~(size_t)255;
    return p;
  };

  _Float16* bufA = (_Float16*)take((size_t)262144 * 128 * 2);
  _Float16* bufB = (_Float16*)take((size_t)262144 * 128 * 2);
  int*   idx1 = (int*)take(4096 * 4);
  float* pos1 = (float*)take(4096 * 3 * 4);
  int*   idx2 = (int*)take(1024 * 4);
  float* pos2 = (float*)take(1024 * 3 * 4);
  int*   nb1  = (int*)take((size_t)4096 * 64 * 4);
  int*   cnt1 = (int*)take(4096 * 4);
  int*   nb2  = (int*)take((size_t)1024 * 64 * 4);
  int*   cnt2 = (int*)take(1024 * 4);
  _Float16* x1   = (_Float16*)take((size_t)4096 * 128 * 2);
  _Float16* x2   = (_Float16*)take((size_t)1024 * 256 * 2);
  _Float16* afc  = (_Float16*)take(16 * 1024 * 2);
  _Float16* yfc1 = (_Float16*)take(16 * 512 * 2);
  _Float16* yfc2 = (_Float16*)take(16 * 256 * 2);

  // d_in layout (setup_inputs dict order): 0=points, 1=features, then params
  // mlp layers contribute {W,b,gamma,beta,mu,var}, fc layers {W,b}.
  struct L { int widx, cout, cin, kp, bn; };
  const L Ls[11] = {
    {2,   64,   4,   32, 1}, {8,   64,  64,   64, 1}, {14, 128,  64,   64, 1},
    {20, 128, 131,  160, 1}, {26, 128, 128,  128, 1}, {32, 256, 128,  128, 1},
    {38, 256, 259,  288, 1}, {44, 512, 256,  256, 1}, {50, 1024, 512, 512, 1},
    {56, 512, 1024, 1024, 0}, {58, 256, 512, 512, 0}
  };
  _Float16* wh[11];
  for (int i = 0; i < 11; ++i) {
    wh[i] = (_Float16*)take((size_t)Ls[i].cout * Ls[i].kp * 2);
    int tot = Ls[i].cout * Ls[i].kp;
    convw_kernel<<<(tot + 255) / 256, 256, 0, stream>>>(
        (const float*)d_in[Ls[i].widx], wh[i], Ls[i].cout, Ls[i].cin, Ls[i].kp);
  }

  auto gemm = [&](const _Float16* A, int li, _Float16* Y, int R) {
    const L& l = Ls[li];
    const float* bb = (const float*)d_in[l.widx + 1];
    const float* gg = l.bn ? (const float*)d_in[l.widx + 2] : nullptr;
    const float* be = l.bn ? (const float*)d_in[l.widx + 3] : nullptr;
    const float* mm = l.bn ? (const float*)d_in[l.widx + 4] : nullptr;
    const float* vv = l.bn ? (const float*)d_in[l.widx + 5] : nullptr;
    dim3 grid(R / 16, l.cout / 16);
    gemm_wmma<<<grid, 32, 0, stream>>>(A, wh[li], bb, gg, be, mm, vv, Y, l.kp, l.cout, l.bn);
  };

  const float* points = (const float*)d_in[0];
  const float* feats  = (const float*)d_in[1];

  // ---- Set abstraction 1: 8192 -> 4096, r = 0.03 ----
  fps_kernel<<<1, 1024, 0, stream>>>(points, 8192, 4096, idx1, pos1);
  radius_kernel<<<4096, 32, 0, stream>>>(points, 8192, pos1, nb1, cnt1, 0.03f * 0.03f);
  gather1_kernel<<<(262144 + 255) / 256, 256, 0, stream>>>(feats, points, nb1, pos1, bufA, 4096);
  gemm(bufA, 0, bufB, 262144);
  gemm(bufB, 1, bufA, 262144);
  gemm(bufA, 2, bufB, 262144);
  maxpool_kernel<<<(4096 * 128 + 255) / 256, 256, 0, stream>>>(bufB, cnt1, x1, 4096, 128);

  // ---- Set abstraction 2: 4096 -> 1024, r = 0.2 ----
  fps_kernel<<<1, 1024, 0, stream>>>(pos1, 4096, 1024, idx2, pos2);
  radius_kernel<<<1024, 32, 0, stream>>>(pos1, 4096, pos2, nb2, cnt2, 0.2f * 0.2f);
  gather2_kernel<<<(65536 + 255) / 256, 256, 0, stream>>>(x1, pos1, nb2, pos2, bufA, 1024);
  gemm(bufA, 3, bufB, 65536);
  gemm(bufB, 4, bufA, 65536);
  gemm(bufA, 5, bufB, 65536);
  maxpool_kernel<<<(1024 * 256 + 255) / 256, 256, 0, stream>>>(bufB, cnt2, x2, 1024, 256);

  // ---- mlp3 + global max + FC head ----
  concat3_kernel<<<(1024 + 255) / 256, 256, 0, stream>>>(x2, pos2, bufA);
  gemm(bufA, 6, bufB, 1024);
  gemm(bufB, 7, bufA, 1024);
  gemm(bufA, 8, bufB, 1024);
  gmax_kernel<<<4, 256, 0, stream>>>(bufB, afc);
  gemm(afc, 9, yfc1, 16);
  gemm(yfc1, 10, yfc2, 16);
  fc3_kernel<<<1, 32, 0, stream>>>(yfc2, (const float*)d_in[60], (const float*)d_in[61],
                                   (float*)d_out);
}
